// RNN_88708254531999
// MI455X (gfx1250) — matrix operand order repaired and checked by reading, not compiled, not run on hardware
//
#include <hip/hip_runtime.h>
#include <hip/hip_bf16.h>

// ---------------- Problem constants (match reference) ----------------
#define S_LEN 128
#define BATCH 64
#define VOCAB 10000
#define VOCAB_PAD 10112   // 79 * 128 (N-tile 128, zero-padded Wy)
#define EMB   512
#define HID   512
#define BH    (BATCH * HID)   // 32768

typedef __attribute__((ext_vector_type(16))) __bf16 v16bf;
typedef __attribute__((ext_vector_type(8)))  float  v8f;

// ---------------- Helpers ----------------
static __device__ __forceinline__ unsigned short f32_to_bf16(float f) {
    union { float f; unsigned int u; } x; x.f = f;
    unsigned int u = x.u;
    unsigned int r = (u + 0x7FFFu + ((u >> 16) & 1u)) >> 16;  // RNE
    return (unsigned short)r;
}
static __device__ __forceinline__ float bf16_to_f32(unsigned short h) {
    union { unsigned int u; float f; } x; x.u = ((unsigned int)h) << 16;
    return x.f;
}

// Async global -> LDS copy, 16B per lane, tracked by ASYNCcnt (CDNA5).
// Flat pointer to __shared__ = {LDS aperture hi32 | lds offset lo32}, so the
// low 32 bits are exactly the wave-relative LDS address the instruction wants.
static __device__ __forceinline__ void async_b128(unsigned lds_off, const void* g) {
    asm volatile("global_load_async_to_lds_b128 %0, %1, off"
                 :: "v"(lds_off), "v"((unsigned long long)(uintptr_t)g)
                 : "memory");
}
static __device__ __forceinline__ void wait_async0() {
    asm volatile("s_wait_asynccnt 0x0" ::: "memory");
}

// ---------------- Elementwise prep kernels ----------------
__global__ void k_cvt_bf16(const float* __restrict__ src,
                           unsigned short* __restrict__ dst, int n) {
    int i = blockIdx.x * blockDim.x + threadIdx.x;
    if (i < n) dst[i] = f32_to_bf16(src[i]);
}

// dst is [rowsPad, cols]; rows >= rows are zero-filled.
__global__ void k_cvt_pad_bf16(const float* __restrict__ src,
                               unsigned short* __restrict__ dst,
                               int rows, int rowsPad, int cols) {
    int i = blockIdx.x * blockDim.x + threadIdx.x;
    if (i < rowsPad * cols) {
        int r = i / cols;
        dst[i] = (r < rows) ? f32_to_bf16(src[i]) : (unsigned short)0;
    }
}

// Bcat[n, k] = k<HID ? W_in[n,k] : W_hh[n,k-HID]   (row-major, ldb = 2*HID)
__global__ void k_build_bcat(const float* __restrict__ Win,
                             const float* __restrict__ Whh,
                             unsigned short* __restrict__ Bcat) {
    int i = blockIdx.x * blockDim.x + threadIdx.x;
    if (i < HID * 2 * HID) {
        int n = i >> 10;
        int k = i & 1023;
        float v = (k < HID) ? Win[n * HID + k] : Whh[n * HID + (k - HID)];
        Bcat[i] = f32_to_bf16(v);
    }
}

__global__ void k_add_vec(const float* __restrict__ a, const float* __restrict__ b,
                          float* __restrict__ out, int n) {
    int i = blockIdx.x * blockDim.x + threadIdx.x;
    if (i < n) out[i] = a[i] + b[i];
}

// E_all[r, e] = bf16(emb[tokens[r], e]),  r in [0, S*B)
__global__ void k_gather_emb(const int* __restrict__ tokens,
                             const float* __restrict__ emb,
                             unsigned short* __restrict__ Eall) {
    int i = blockIdx.x * blockDim.x + threadIdx.x;
    if (i < S_LEN * BATCH * EMB) {
        int r = i / EMB;
        int e = i - r * EMB;
        int tok = tokens[r];
        Eall[i] = f32_to_bf16(emb[(size_t)tok * EMB + e]);
    }
}

__global__ void k_init_state(const float* __restrict__ hidden,
                             unsigned short* __restrict__ H0,
                             unsigned short* __restrict__ H1) {
    int i = blockIdx.x * blockDim.x + threadIdx.x;
    if (i < BH) {
        H0[i] = f32_to_bf16(hidden[i]);
        H1[i] = f32_to_bf16(hidden[BH + i]);
    }
}

__global__ void k_final_hidden(const unsigned short* __restrict__ H0,
                               const unsigned short* __restrict__ H1,
                               float* __restrict__ out) {
    int i = blockIdx.x * blockDim.x + threadIdx.x;
    if (i < BH) {
        out[i]      = bf16_to_f32(H0[i]);
        out[BH + i] = bf16_to_f32(H1[i]);
    }
}

// ---------------- Generic NT WMMA GEMM (async staged, double buffered) ----
// C[M,N] = act( A[M,K](bf16) * Bw[N,K]^T(bf16)  [+ bias[n]] [+ src[m,n]] )
// A rows: k < kSplit -> A [lda], else A2 [lda2] at column (k-kSplit).
// 128 threads = 4 wave32 in a 2(M) x 2(N) layout; WG tile 64(M) x 128(N);
// each wave computes 32x64 via 2 A-frags x 4 B-frags = 8 WMMA per K-chunk.
// M, K multiples of 64/32. B rows [nBase, nBase+128) must be readable
// (caller pads weights); C stores are guarded by n < N.
template<bool ADD_BIAS, bool ADD_SRC, bool DO_TANH, bool OUT_F32>
__global__ __launch_bounds__(128)
void k_gemm_nt(const unsigned short* __restrict__ A,  int lda,
               const unsigned short* __restrict__ A2, int lda2, int kSplit,
               const unsigned short* __restrict__ Bw, int ldb,
               float* __restrict__ Cf, unsigned short* __restrict__ Cb, int ldc,
               const float* __restrict__ bias,
               const float* __restrict__ src, int ldsrc,
               int N, int K) {
    constexpr int LSTR = 40;   // LDS row stride (elems): 80B keeps 16B alignment
    __shared__ __attribute__((aligned(16))) unsigned short As[2][64  * LSTR];
    __shared__ __attribute__((aligned(16))) unsigned short Bs[2][128 * LSTR];

    const int tid   = threadIdx.x;
    const int wave  = tid >> 5;
    const int lane  = tid & 31;
    const int waveM = wave & 1;          // 2 waves over M
    const int waveN = wave >> 1;         // 2 waves over N
    const int mBase = blockIdx.x * 64;
    const int nBase = blockIdx.y * 128;

    const int aRow = tid >> 1;           // 0..63
    const int aCol = (tid & 1) * 16;     // elem offset in 32-wide K chunk

    const int laneM = lane & 15;
    const int hi    = lane >> 4;         // half-wave select
    const int kbA   = hi * 16;           // A frag: K 0-15 vs 16-31
    const int kbB   = hi * 16;           // B frag: K 0-15 vs 16-31

    v8f c[2][4];
    #pragma unroll
    for (int mf = 0; mf < 2; ++mf)
        #pragma unroll
        for (int j = 0; j < 4; ++j) c[mf][j] = v8f{};

    auto issue = [&](int kc, int buf) {
        // A tile [64 x 32]: 32B per thread (2 async b128)
        {
            const int kg = kc + aCol;
            const unsigned short* ap =
                (kg < kSplit) ? (A  + (size_t)(mBase + aRow) * lda  + kg)
                              : (A2 + (size_t)(mBase + aRow) * lda2 + (kg - kSplit));
            const unsigned s = (unsigned)(uintptr_t)&As[buf][aRow * LSTR + aCol];
            async_b128(s,      ap);
            async_b128(s + 16, ap + 8);
        }
        // B tile [128 x 32]: 64B per thread, one full row (4 async b128)
        {
            const unsigned short* bp = Bw + (size_t)(nBase + tid) * ldb + kc;
            const unsigned s = (unsigned)(uintptr_t)&Bs[buf][tid * LSTR];
            async_b128(s,      bp);
            async_b128(s + 16, bp + 8);
            async_b128(s + 32, bp + 16);
            async_b128(s + 48, bp + 24);
        }
    };

    issue(0, 0);
    const int nChunks = K >> 5;
    for (int it = 0; it < nChunks; ++it) {
        const int buf = it & 1;
        wait_async0();        // this wave's chunk `it` landed in LDS
        __syncthreads();      // all waves' chunk `it` landed; buf^1 free to reuse
        if (it + 1 < nChunks) issue((it + 1) << 5, buf ^ 1);  // overlap with compute

        union { v16bf v; uint4 q[2]; } af[2];
        #pragma unroll
        for (int mf = 0; mf < 2; ++mf) {
            const int row = waveM * 32 + mf * 16 + laneM;
            af[mf].q[0] = *(const uint4*)&As[buf][row * LSTR + kbA];
            af[mf].q[1] = *(const uint4*)&As[buf][row * LSTR + kbA + 8];
        }
        #pragma unroll
        for (int j = 0; j < 4; ++j) {
            union { v16bf v; uint4 q[2]; } bfr;
            const int row = waveN * 64 + j * 16 + laneM;
            const uint4* p = (const uint4*)&Bs[buf][row * LSTR + kbB];
            bfr.q[0] = p[0]; bfr.q[1] = p[1];
            #pragma unroll
            for (int mf = 0; mf < 2; ++mf)
                c[mf][j] = __builtin_amdgcn_wmma_f32_16x16x32_bf16(
                    false, af[mf].v, false, bfr.v, (short)0, c[mf][j], false, false);
        }
    }

    // epilogue: f32 C/D layout -> lane (laneM, hi), VGPR v -> row v + 8*hi
    #pragma unroll
    for (int mf = 0; mf < 2; ++mf) {
        #pragma unroll
        for (int j = 0; j < 4; ++j) {
            #pragma unroll
            for (int v = 0; v < 8; ++v) {
                const int m = mBase + waveM * 32 + mf * 16 + v + hi * 8;
                const int n = nBase + waveN * 64 + j * 16 + laneM;
                if (n < N) {
                    float val = c[mf][j][v];
                    if (ADD_BIAS) val += bias[n];
                    if (ADD_SRC)  val += src[(size_t)m * ldsrc + n];
                    if (DO_TANH)  val = tanhf(val);
                    if (OUT_F32)  Cf[(size_t)m * ldc + n] = val;
                    else          Cb[(size_t)m * ldc + n] = f32_to_bf16(val);
                }
            }
        }
    }
}

// ---------------- Host orchestration ----------------
extern "C" void kernel_launch(void* const* d_in, const int* in_sizes, int n_in,
                              void* d_out, int out_size, void* d_ws, size_t ws_size,
                              hipStream_t stream) {
    const int*   inputs = (const int*)  d_in[0];
    const float* hidden = (const float*)d_in[1];
    const float* emb    = (const float*)d_in[2];
    const float* W_ih0  = (const float*)d_in[3];
    const float* b_ih0  = (const float*)d_in[4];
    const float* W_hh0  = (const float*)d_in[5];
    const float* b_hh0  = (const float*)d_in[6];
    const float* W_in   = (const float*)d_in[7];
    const float* b_in   = (const float*)d_in[8];
    const float* W_hh   = (const float*)d_in[9];
    const float* b_hh   = (const float*)d_in[10];
    const float* Wy     = (const float*)d_in[11];
    const float* by     = (const float*)d_in[12];
    float* out = (float*)d_out;

    // workspace carve-up (256B aligned)
    char* ws = (char*)d_ws;
    size_t off = 0;
    auto carve = [&](size_t bytes) -> void* {
        void* p = ws + off;
        off = (off + bytes + 255) & ~(size_t)255;
        return p;
    };
    unsigned short* Whh0b = (unsigned short*)carve((size_t)HID * HID * 2);
    unsigned short* Wih0b = (unsigned short*)carve((size_t)HID * EMB * 2);
    unsigned short* Bcat  = (unsigned short*)carve((size_t)HID * 2 * HID * 2);
    unsigned short* Wyb   = (unsigned short*)carve((size_t)VOCAB_PAD * HID * 2);
    float*          bias0 = (float*)carve((size_t)HID * 4);
    float*          bias1 = (float*)carve((size_t)HID * 4);
    unsigned short* Eall  = (unsigned short*)carve((size_t)S_LEN * BATCH * EMB * 2);
    float*          X0    = (float*)carve((size_t)S_LEN * BATCH * HID * 4);
    unsigned short* H0    = (unsigned short*)carve((size_t)(S_LEN + 1) * BH * 2);
    unsigned short* H1    = (unsigned short*)carve((size_t)(S_LEN + 1) * BH * 2);

    const int TB = 256;
    auto gsz = [](long long n, int tb) { return (unsigned)((n + tb - 1) / tb); };

    // -------- phase 0: precision prep --------
    k_cvt_bf16<<<gsz((long long)HID * HID, TB), TB, 0, stream>>>(W_hh0, Whh0b, HID * HID);
    k_cvt_bf16<<<gsz((long long)HID * EMB, TB), TB, 0, stream>>>(W_ih0, Wih0b, HID * EMB);
    k_cvt_pad_bf16<<<gsz((long long)VOCAB_PAD * HID, TB), TB, 0, stream>>>(
        Wy, Wyb, VOCAB, VOCAB_PAD, HID);
    k_build_bcat<<<gsz((long long)HID * 2 * HID, TB), TB, 0, stream>>>(W_in, W_hh, Bcat);
    k_add_vec<<<gsz(HID, TB), TB, 0, stream>>>(b_ih0, b_hh0, bias0, HID);
    k_add_vec<<<gsz(HID, TB), TB, 0, stream>>>(b_in, b_hh, bias1, HID);
    k_gather_emb<<<gsz((long long)S_LEN * BATCH * EMB, TB), TB, 0, stream>>>(inputs, emb, Eall);
    k_init_state<<<gsz(BH, TB), TB, 0, stream>>>(hidden, H0, H1);

    // -------- phase 1: X0 = E_all @ W_ih0^T + (b_ih0 + b_hh0)   [8192 x 512] --------
    {
        dim3 grid(S_LEN * BATCH / 64, HID / 128);
        k_gemm_nt<true, false, false, true><<<grid, 128, 0, stream>>>(
            Eall, EMB, Eall, EMB, EMB, Wih0b, EMB,
            X0, nullptr, HID, bias0, nullptr, 0, HID, EMB);
    }

    // -------- phase 2: serial recurrence --------
    for (int t = 0; t < S_LEN; ++t) {
        // h0^t = tanh( X0[t] + h0^{t-1} @ W_hh0^T )
        {
            dim3 grid(1, HID / 128);
            k_gemm_nt<false, true, true, false><<<grid, 128, 0, stream>>>(
                H0 + (size_t)t * BH, HID, H0 + (size_t)t * BH, HID, HID,
                Whh0b, HID,
                nullptr, H0 + (size_t)(t + 1) * BH, HID,
                nullptr, X0 + (size_t)t * BH, HID, HID, HID);
        }
        // h1^t = tanh( [h0^t | h1^{t-1}] @ [W_in ; W_hh]^T + (b_in + b_hh) ), K = 1024
        {
            dim3 grid(1, HID / 128);
            k_gemm_nt<true, false, true, false><<<grid, 128, 0, stream>>>(
                H0 + (size_t)(t + 1) * BH, HID, H1 + (size_t)t * BH, HID, HID,
                Bcat, 2 * HID,
                nullptr, H1 + (size_t)(t + 1) * BH, HID,
                bias1, nullptr, 0, HID, 2 * HID);
        }
    }

    // -------- phase 3: logits = H1_all @ Wy^T + by   [8192 x 10000] --------
    {
        dim3 grid(S_LEN * BATCH / 64, (VOCAB + 127) / 128);
        k_gemm_nt<true, false, false, true><<<grid, 128, 0, stream>>>(
            H1 + BH, HID, H1 + BH, HID, HID, Wyb, HID,
            out, nullptr, VOCAB, by, nullptr, 0, VOCAB, HID);
    }

    // -------- phase 4: hidden_final (bf16 -> f32) into output tail --------
    k_final_hidden<<<gsz(BH, TB), TB, 0, stream>>>(
        H0 + (size_t)S_LEN * BH, H1 + (size_t)S_LEN * BH,
        out + (size_t)S_LEN * BATCH * VOCAB);
}
